// BiLSTM_CRF_55611236548713
// MI455X (gfx1250) — matrix operand order repaired and checked by reading, not compiled, not run on hardware
//
#include <hip/hip_runtime.h>
#include <hip/hip_bf16.h>
#include <math.h>

// ---------------- model dims ----------------
#define BB   32
#define TT   256
#define LL   16
#define EE   300
#define ECC  50
#define HH   512
#define G4H  2048          // 4*H
#define INW  400           // E + 2*EC
#define INP  416           // padded to 13*32
#define KT   48
#define TAG_START 46
#define TAG_STOP  47
#define SEQ  (BB*TT)       // 8192

// ---------------- WMMA plumbing ----------------
typedef __attribute__((ext_vector_type(16))) __bf16          v16bf;
typedef __attribute__((ext_vector_type(16))) unsigned short  v16us;
typedef __attribute__((ext_vector_type(8)))  float           v8f;

__device__ inline unsigned short f2bf(float f) {
    unsigned int u = __float_as_uint(f);
    u += 0x7FFFu + ((u >> 16) & 1u);       // round-to-nearest-even
    return (unsigned short)(u >> 16);
}

__device__ inline v8f zero8() {
    v8f z;
#pragma unroll
    for (int i = 0; i < 8; ++i) z[i] = 0.0f;
    return z;
}

// A fragment: 16x32 bf16, row-major source, layout per ISA 7.12.2
__device__ inline v16bf frag_A(const unsigned short* A, int lda, int m0, int k0) {
    int lane = threadIdx.x & 31;
    int hf   = lane >> 4;
    const unsigned short* row = A + (size_t)(m0 + (lane & 15)) * lda + k0 + hf * 8;
    v16us u;
#pragma unroll
    for (int e = 0; e < 16; ++e) {
        int k = (e & 7) + ((e >> 3) << 4);   // {0..7} then {16..23}
        u[e] = row[k];
    }
    return __builtin_bit_cast(v16bf, u);
}

// B fragment: B = W^T (W stored row-major [N x K]); B is 32x16
__device__ inline v16bf frag_BT(const unsigned short* W, int ldw, int n0, int k0) {
    int lane = threadIdx.x & 31;
    int hf   = lane >> 4;
    const unsigned short* row = W + (size_t)(n0 + (lane & 15)) * ldw + k0 + hf * 8;
    v16us u;
#pragma unroll
    for (int e = 0; e < 16; ++e) u[e] = row[(e & 7) + ((e >> 3) << 4)];
    return __builtin_bit_cast(v16bf, u);
}

__device__ inline v8f wmma_bf16(v16bf a, v16bf b, v8f c) {
    return __builtin_amdgcn_wmma_f32_16x16x32_bf16(
        /*neg_a=*/false, a, /*neg_b=*/false, b,
        /*c_mod=*/(short)0, c, /*reuse_a=*/false, /*reuse_b=*/false);
}

__device__ inline float sigm(float x) { return 1.0f / (1.0f + expf(-x)); }

// ---------------- weight packing fp32 -> padded bf16 ----------------
__global__ void k_pack_bf16(const float* __restrict__ src, unsigned short* __restrict__ dst,
                            int N, int K, int Kp) {
    int idx = blockIdx.x * blockDim.x + threadIdx.x;
    if (idx >= N * Kp) return;
    int n = idx / Kp, k = idx % Kp;
    float v = (k < K) ? src[(size_t)n * K + k] : 0.0f;
    dst[idx] = f2bf(v);
}

// ---------------- char BiLSTM: one wave32 per sequence ----------------
// dir = block-uniform (blocks 0..1023 fwd, 1024..2047 bwd); 8 waves/block
__global__ void k_char_lstm(const int* __restrict__ char_ids, const float* __restrict__ char_emb,
                            const float* __restrict__ Wih_f, const float* __restrict__ Whh_f,
                            const float* __restrict__ b_f,
                            const float* __restrict__ Wih_b, const float* __restrict__ Whh_b,
                            const float* __restrict__ b_b,
                            float* __restrict__ cfeat) {
    int wv   = (blockIdx.x * blockDim.x + threadIdx.x) >> 5;
    int lane = threadIdx.x & 31;
    int wl   = threadIdx.x >> 5;
    int dir  = wv >> 13;            // 8192 waves per direction
    int seq  = wv & 8191;
    const float* Wih  = dir ? Wih_b : Wih_f;
    const float* Whh  = dir ? Whh_b : Whh_f;
    const float* bias = dir ? b_b   : b_f;

    // per-wave LDS: h[50] c[50] x[50] z[200]
    __shared__ float sh[8][352];
    float* h  = sh[wl];
    float* c  = h + 50;
    float* xv = c + 50;
    float* zl = xv + 50;
    for (int i = lane; i < 50; i += 32) { h[i] = 0.0f; c[i] = 0.0f; }
    __syncthreads();

    for (int s = 0; s < LL; ++s) {
        int t = dir ? (LL - 1 - s) : s;
        int cid = char_ids[(size_t)seq * LL + t];
        const float* e = char_emb + (size_t)cid * ECC;
        for (int i = lane; i < 50; i += 32) xv[i] = e[i];
        __syncthreads();
        // gates z[g] = b[g] + Wih[g,:]*x + Whh[g,:]*h   (200 gates / 32 lanes)
        for (int j = 0; j < 7; ++j) {
            int g = lane + 32 * j;
            if (g < 200) {
                float acc = bias[g];
                const float* wi = Wih + (size_t)g * ECC;
                const float* wh = Whh + (size_t)g * ECC;
#pragma unroll 10
                for (int k = 0; k < ECC; ++k) acc += wi[k] * xv[k];
#pragma unroll 10
                for (int k = 0; k < ECC; ++k) acc += wh[k] * h[k];
                zl[g] = acc;
            }
        }
        __syncthreads();
        for (int u = lane; u < ECC; u += 32) {
            float zi = zl[u], zf = zl[50 + u], zg = zl[100 + u], zo = zl[150 + u];
            float cc = sigm(zf) * c[u] + sigm(zi) * tanhf(zg);
            c[u] = cc;
            h[u] = sigm(zo) * tanhf(cc);
        }
        __syncthreads();
        if (dir == 0 && s == LL - 1)     // fwd final h
            for (int u = lane; u < ECC; u += 32) cfeat[(size_t)seq * (2 * ECC) + u] = h[u];
        if (dir == 1 && s == 0)          // hb[-1] = reverse h after first step (time L-1)
            for (int u = lane; u < ECC; u += 32) cfeat[(size_t)seq * (2 * ECC) + ECC + u] = h[u];
        __syncthreads();
    }
}

// ---------------- build x_bf16 = [word_emb | cfeat | 0-pad] ----------------
__global__ void k_build_x(const int* __restrict__ word_ids, const float* __restrict__ word_emb,
                          const float* __restrict__ cfeat, unsigned short* __restrict__ xbf) {
    int idx = blockIdx.x * blockDim.x + threadIdx.x;
    if (idx >= SEQ * INP) return;
    int seq = idx / INP, col = idx % INP;
    float v;
    if (col < EE)          v = word_emb[(size_t)word_ids[seq] * EE + col];
    else if (col < INW)    v = cfeat[(size_t)seq * (2 * ECC) + (col - EE)];
    else                   v = 0.0f;
    xbf[idx] = f2bf(v);
}

// ---------------- input projection GEMM: xi = x @ Wih^T + b ----------------
// M=8192, N=2048, K=416. One wave -> 16x64 tile (4 accumulators, A reuse).
__global__ void k_input_gemm(const unsigned short* __restrict__ xbf,
                             const unsigned short* __restrict__ wihp,
                             const float* __restrict__ bias, float* __restrict__ xi) {
    int wave = (blockIdx.x * blockDim.x + threadIdx.x) >> 5;
    int nt = wave & 31;            // 32 tiles of 64 cols
    int mt = wave >> 5;            // 512 tiles of 16 rows
    if (mt >= SEQ / 16) return;    // wave-uniform
    int m0 = mt * 16, n0 = nt * 64;
    v8f acc[4] = { zero8(), zero8(), zero8(), zero8() };
    for (int k0 = 0; k0 < INP; k0 += 32) {
        v16bf a = frag_A(xbf, INP, m0, k0);
#pragma unroll
        for (int j = 0; j < 4; ++j) {
            v16bf b = frag_BT(wihp, INP, n0 + j * 16, k0);
            acc[j] = wmma_bf16(a, b, acc[j]);
        }
    }
    int lane = threadIdx.x & 31, hf = lane >> 4, nl = lane & 15;
#pragma unroll
    for (int j = 0; j < 4; ++j) {
        float bv = bias[n0 + j * 16 + nl];
#pragma unroll
        for (int r = 0; r < 8; ++r)
            xi[(size_t)(m0 + r + hf * 8) * G4H + n0 + j * 16 + nl] = acc[j][r] + bv;
    }
}

// ---------------- init recurrent state ----------------
__global__ void k_init_state(unsigned short* __restrict__ hbf, float* __restrict__ cst) {
    int idx = blockIdx.x * blockDim.x + threadIdx.x;
    if (idx >= 2 * BB * HH) return;
    hbf[idx] = 0;               // bf16 zero
    cst[idx] = 0.0f;
}

// ---------------- per-step recurrent GEMM: z = h @ Whh^T + xi[:,t,:] ----------------
// both directions; M=32 (2 tiles), N=2048 (32 tiles of 64), K=512; 128 waves.
__global__ void k_recur_gemm(const unsigned short* __restrict__ hbf,
                             const unsigned short* __restrict__ whhp_f,
                             const unsigned short* __restrict__ whhp_b,
                             const float* __restrict__ xi_f, const float* __restrict__ xi_b,
                             float* __restrict__ z, int t) {
    int wave = (blockIdx.x * blockDim.x + threadIdx.x) >> 5;
    int nt  = wave & 31;
    int mt  = (wave >> 5) & 1;
    int dir = wave >> 6;
    const unsigned short* W  = dir ? whhp_b : whhp_f;
    const float*          xi = dir ? xi_b  : xi_f;
    const unsigned short* h  = hbf + (size_t)dir * BB * HH;
    int tt = dir ? (TT - 1 - t) : t;
    int m0 = mt * 16, n0 = nt * 64;
    v8f acc[4] = { zero8(), zero8(), zero8(), zero8() };
    for (int k0 = 0; k0 < HH; k0 += 32) {
        v16bf a = frag_A(h, HH, m0, k0);
#pragma unroll
        for (int j = 0; j < 4; ++j) {
            v16bf b = frag_BT(W, HH, n0 + j * 16, k0);
            acc[j] = wmma_bf16(a, b, acc[j]);
        }
    }
    int lane = threadIdx.x & 31, hf = lane >> 4, nl = lane & 15;
#pragma unroll
    for (int j = 0; j < 4; ++j)
#pragma unroll
        for (int r = 0; r < 8; ++r) {
            int m = m0 + r + hf * 8;                       // batch row
            int n = n0 + j * 16 + nl;
            z[(size_t)dir * BB * G4H + (size_t)m * G4H + n] =
                acc[j][r] + xi[((size_t)m * TT + tt) * G4H + n];
        }
}

// ---------------- per-step gate math + h/c update ----------------
__global__ void k_lstm_point(const float* __restrict__ z, float* __restrict__ cst,
                             unsigned short* __restrict__ hbf,
                             unsigned short* __restrict__ outbf, int t) {
    int idx = blockIdx.x * blockDim.x + threadIdx.x;    // 2*32*512
    if (idx >= 2 * BB * HH) return;
    int hidx = idx & (HH - 1);
    int b    = (idx >> 9) & (BB - 1);
    int dir  = idx >> 14;
    const float* zd = z + (size_t)dir * BB * G4H + (size_t)b * G4H;
    float zi = zd[hidx], zf = zd[HH + hidx], zg = zd[2 * HH + hidx], zo = zd[3 * HH + hidx];
    float c = cst[idx];
    c = sigm(zf) * c + sigm(zi) * tanhf(zg);
    float h = sigm(zo) * tanhf(c);
    cst[idx] = c;
    unsigned short hb = f2bf(h);
    hbf[idx] = hb;                                       // A-matrix source for next step
    int tt = dir ? (TT - 1 - t) : t;
    outbf[((size_t)b * TT + tt) * (2 * HH) + (size_t)dir * HH + hidx] = hb;
}

// ---------------- tag projection: feats = out @ Wtag^T + btag ----------------
// M=8192 (512 tiles), N=48 (3 tiles), K=1024.
__global__ void k_tag_gemm(const unsigned short* __restrict__ outbf,
                           const unsigned short* __restrict__ wtagp,
                           const float* __restrict__ btag, float* __restrict__ feats) {
    int wave = (blockIdx.x * blockDim.x + threadIdx.x) >> 5;
    int nt = wave % 3;
    int mt = wave / 3;
    if (mt >= SEQ / 16) return;    // wave-uniform
    int m0 = mt * 16, n0 = nt * 16;
    v8f acc = zero8();
    for (int k0 = 0; k0 < 2 * HH; k0 += 32) {
        v16bf a = frag_A(outbf, 2 * HH, m0, k0);
        v16bf b = frag_BT(wtagp, 2 * HH, n0, k0);
        acc = wmma_bf16(a, b, acc);
    }
    int lane = threadIdx.x & 31, hf = lane >> 4, nl = lane & 15;
    float bv = btag[n0 + nl];
#pragma unroll
    for (int r = 0; r < 8; ++r)
        feats[(size_t)(m0 + r + hf * 8) * KT + n0 + nl] = acc[r] + bv;
}

// ---------------- Viterbi decode: one workgroup per batch row ----------------
__global__ void k_viterbi(const float* __restrict__ feats, const float* __restrict__ trans,
                          int* __restrict__ bp_ws, float* __restrict__ out) {
    int b = blockIdx.x;
    int tid = threadIdx.x;                 // 64 threads
    __shared__ float fv[KT], fvn[KT];
    __shared__ float str[KT * KT];
    for (int i = tid; i < KT * KT; i += 64) str[i] = trans[i];
    if (tid < KT) fv[tid] = (tid == TAG_START) ? 0.0f : -10000.0f;
    __syncthreads();

    int*        bp = bp_ws + (size_t)b * TT * KT;
    const float* fb = feats + (size_t)b * TT * KT;
    for (int t = 0; t < TT; ++t) {
        if (tid < KT) {
            float best = -3.4e38f; int bi = 0;
            for (int p = 0; p < KT; ++p) {
                float s = fv[p] + str[tid * KT + p];
                if (s > best) { best = s; bi = p; }
            }
            fvn[tid] = best + fb[(size_t)t * KT + tid];
            bp[(size_t)t * KT + tid] = bi;
        }
        __syncthreads();
        if (tid < KT) fv[tid] = fvn[tid];
        __syncthreads();
    }
    if (tid == 0) {
        float best = -3.4e38f; int bi = 0;
        for (int p = 0; p < KT; ++p) {
            float s = fv[p] + str[TAG_STOP * KT + p];
            if (s > best) { best = s; bi = p; }
        }
        out[b] = best;                          // score
        float* path = out + BB + (size_t)b * TT;
        int tag = bi;
        for (int t = TT - 1; t >= 0; --t) {
            path[t] = (float)tag;
            tag = bp[(size_t)t * KT + tag];
        }
    }
}

// ---------------- host orchestration ----------------
extern "C" void kernel_launch(void* const* d_in, const int* in_sizes, int n_in,
                              void* d_out, int out_size, void* d_ws, size_t ws_size,
                              hipStream_t stream) {
    (void)in_sizes; (void)n_in; (void)out_size; (void)ws_size;
    const int*   word_ids = (const int*)  d_in[0];
    const int*   char_ids = (const int*)  d_in[1];
    const float* word_emb = (const float*)d_in[2];
    const float* char_emb = (const float*)d_in[3];
    const float* cWih_f   = (const float*)d_in[4];
    const float* cWhh_f   = (const float*)d_in[5];
    const float* cb_f     = (const float*)d_in[6];
    const float* cWih_b   = (const float*)d_in[7];
    const float* cWhh_b   = (const float*)d_in[8];
    const float* cb_b     = (const float*)d_in[9];
    const float* Wih_f    = (const float*)d_in[10];
    const float* Whh_f    = (const float*)d_in[11];
    const float* b_f      = (const float*)d_in[12];
    const float* Wih_b    = (const float*)d_in[13];
    const float* Whh_b    = (const float*)d_in[14];
    const float* b_b      = (const float*)d_in[15];
    const float* Wtag     = (const float*)d_in[16];
    const float* btag     = (const float*)d_in[17];
    const float* trans    = (const float*)d_in[18];

    // ---- workspace carve (256B aligned) ----
    char*  ws  = (char*)d_ws;
    size_t off = 0;
    auto carve = [&](size_t bytes) -> void* {
        void* p = ws + off;
        off = (off + bytes + 255) & ~(size_t)255;
        return p;
    };
    unsigned short* wihp_f = (unsigned short*)carve((size_t)G4H * INP * 2);
    unsigned short* wihp_b = (unsigned short*)carve((size_t)G4H * INP * 2);
    unsigned short* whhp_f = (unsigned short*)carve((size_t)G4H * HH * 2);
    unsigned short* whhp_b = (unsigned short*)carve((size_t)G4H * HH * 2);
    unsigned short* wtagp  = (unsigned short*)carve((size_t)KT * 2 * HH * 2);
    float*          cfeat  = (float*)        carve((size_t)SEQ * 2 * ECC * 4);
    unsigned short* xbf    = (unsigned short*)carve((size_t)SEQ * INP * 2);
    float*          xi_f   = (float*)        carve((size_t)SEQ * G4H * 4);
    float*          xi_b   = (float*)        carve((size_t)SEQ * G4H * 4);
    unsigned short* hbf    = (unsigned short*)carve((size_t)2 * BB * HH * 2);
    float*          cst    = (float*)        carve((size_t)2 * BB * HH * 4);
    float*          zbuf   = (float*)        carve((size_t)2 * BB * G4H * 4);
    unsigned short* outbf  = (unsigned short*)carve((size_t)SEQ * 2 * HH * 2);
    float*          feats  = (float*)        carve((size_t)SEQ * KT * 4);
    int*            bp     = (int*)          carve((size_t)BB * TT * KT * 4);

    // 1. pack weights to bf16 (pad Wih K 400->416)
    k_pack_bf16<<<(G4H * INP + 255) / 256, 256, 0, stream>>>(Wih_f, wihp_f, G4H, INW, INP);
    k_pack_bf16<<<(G4H * INP + 255) / 256, 256, 0, stream>>>(Wih_b, wihp_b, G4H, INW, INP);
    k_pack_bf16<<<(G4H * HH  + 255) / 256, 256, 0, stream>>>(Whh_f, whhp_f, G4H, HH, HH);
    k_pack_bf16<<<(G4H * HH  + 255) / 256, 256, 0, stream>>>(Whh_b, whhp_b, G4H, HH, HH);
    k_pack_bf16<<<(KT * 2 * HH + 255) / 256, 256, 0, stream>>>(Wtag, wtagp, KT, 2 * HH, 2 * HH);

    // 2. char BiLSTM  (16384 waves, 8/block, dir block-uniform)
    k_char_lstm<<<2048, 256, 0, stream>>>(char_ids, char_emb,
                                          cWih_f, cWhh_f, cb_f,
                                          cWih_b, cWhh_b, cb_b, cfeat);

    // 3. concat embeddings -> bf16 padded input
    k_build_x<<<(SEQ * INP + 255) / 256, 256, 0, stream>>>(word_ids, word_emb, cfeat, xbf);

    // 4. input projections (hoisted out of time loop) — big WMMA GEMMs
    k_input_gemm<<<2048, 256, 0, stream>>>(xbf, wihp_f, b_f, xi_f);
    k_input_gemm<<<2048, 256, 0, stream>>>(xbf, wihp_b, b_b, xi_b);

    // 5. zero recurrent state
    k_init_state<<<(2 * BB * HH + 255) / 256, 256, 0, stream>>>(hbf, cst);

    // 6. time loop: L2-resident recurrent WMMA GEMM + fused gates (both dirs)
    for (int t = 0; t < TT; ++t) {
        k_recur_gemm<<<16, 256, 0, stream>>>(hbf, whhp_f, whhp_b, xi_f, xi_b, zbuf, t);
        k_lstm_point<<<(2 * BB * HH + 255) / 256, 256, 0, stream>>>(zbuf, cst, hbf, outbf, t);
    }

    // 7. tag projection
    k_tag_gemm<<<192, 256, 0, stream>>>(outbf, wtagp, btag, feats);

    // 8. Viterbi decode -> d_out = [score(32) | path(32*256)]
    k_viterbi<<<BB, 64, 0, stream>>>(feats, trans, bp, (float*)d_out);
}
